// ModuleCorrelation_47141561041513
// MI455X (gfx1250) — matrix operand order repaired and checked by reading, not compiled, not run on hardware
//
#include <hip/hip_runtime.h>

typedef _Float16 v16h __attribute__((ext_vector_type(16)));
typedef _Float16 v8h  __attribute__((ext_vector_type(8)));
typedef _Float16 v2h  __attribute__((ext_vector_type(2)));
typedef float    v8f  __attribute__((ext_vector_type(8)));

#define B_   4
#define C_   128
#define H_   128
#define W_   256
#define HW_  (H_ * W_)
#define PAD_ 4
#define SECPX 80          // staged second pixels per block: x0-4 .. x0+75
#define SSTR  136         // LDS stride (halfs) per pixel: 272B, 16B-aligned, ~2-way banks
#define XTILE 64          // x pixels per block (4 waves x 16)
#define NW    4

__global__ __launch_bounds__(128)
void corr81_wmma_kernel(const float* __restrict__ first,
                        const float* __restrict__ second,
                        float* __restrict__ out)
{
    __shared__ __align__(32) _Float16 lds_sec[2][SECPX * SSTR]; // 2*80*136*2 = 43520 B
    __shared__ float lds_out[NW][16 * 36];                      // 9216 B

    const int tid  = threadIdx.x;
    const int wave = tid >> 5;
    const int lane = tid & 31;
    const int m    = lane & 15;   // pixel / column within 16-tile
    const int hi   = lane >> 4;   // lane half (K-split selector)

    const int x0 = blockIdx.x * XTILE;
    const int y  = blockIdx.y;
    const int b  = blockIdx.z;
    const int xw = x0 + wave * 16;   // this wave's M-tile base

    // ---------------- A (first) -> registers, f32->f16, documented A layout ---------------
    // 16-bit A 16x32: lanes0-15: VGPR r<4 -> K=2r,2r+1 ; r>=4 -> K=16+2(r-4),+1 ; lanes16-31: +8
    v16h a[4];
    {
        const float* fb = first + ((size_t)b * C_) * HW_ + (size_t)y * W_ + (xw + m);
        #pragma unroll
        for (int s = 0; s < 4; ++s) {
            #pragma unroll
            for (int r = 0; r < 8; ++r) {
                const int c = 32 * s + (r < 4 ? 2 * r : 16 + 2 * (r - 4)) + 8 * hi;
                const float f0 = fb[(size_t)c * HW_];
                const float f1 = fb[(size_t)(c + 1) * HW_];
                a[s][2 * r]     = (_Float16)f0;
                a[s][2 * r + 1] = (_Float16)f1;
            }
        }
    }

    // ---------------- cooperative staging of one second row into LDS (pixel-major f16) ----
    auto stage = [&](int buf, int dy) {
        const int ys = y + dy;
        const bool rowok = (ys >= 0) && (ys < H_);
        const float* sb = second + ((size_t)b * C_) * HW_ + (size_t)ys * W_;
        for (int i = tid; i < SECPX * (C_ / 2); i += 128) {
            const int cp = i / SECPX;          // channel pair 0..63
            const int px = i % SECPX;          // 0..79  -> x = x0-4+px
            const int xs = x0 - PAD_ + px;
            float f0 = 0.0f, f1 = 0.0f;
            if (rowok && xs >= 0 && xs < W_) {
                f0 = sb[(size_t)(2 * cp) * HW_ + xs];
                f1 = sb[(size_t)(2 * cp + 1) * HW_ + xs];
            }
            v2h h;
            h.x = (_Float16)f0;
            h.y = (_Float16)f1;
            *(v2h*)&lds_sec[buf][px * SSTR + 2 * cp] = h;   // 4B-aligned
        }
    };

    stage(0, -PAD_);

    float* wout = &lds_out[wave][0];
    const float inv_c = 1.0f / (float)C_;

    for (int it = 0; it < 9; ++it) {
        __syncthreads();                       // buffer (it&1) staged
        if (it < 8) stage((it + 1) & 1, it + 1 - PAD_);   // overlap next row with compute
        const int buf = it & 1;
        const int dy  = it - PAD_;

        // -------- two 16x16 N-tiles covering second columns xw-4 .. xw+27 --------
        v8f acc0 = {};
        v8f acc1 = {};
        #pragma unroll
        for (int s = 0; s < 4; ++s) {
            // B 32x16 layout: lane gives N; lanes0-15 hold K=0..7,16..23, lanes16-31 K=8..15,24..31
            const int cb = 32 * s + 8 * hi;
            const _Float16* p0 = &lds_sec[buf][(wave * 16 + m) * SSTR + cb];  // N-tile 0
            const _Float16* p1 = p0 + 16 * SSTR;                              // N-tile 1
            const v8h lo0 = *(const v8h*)p0;
            const v8h hi0 = *(const v8h*)(p0 + 16);
            const v8h lo1 = *(const v8h*)p1;
            const v8h hi1 = *(const v8h*)(p1 + 16);
            const v16h b0 = __builtin_shufflevector(lo0, hi0, 0,1,2,3,4,5,6,7,8,9,10,11,12,13,14,15);
            const v16h b1 = __builtin_shufflevector(lo1, hi1, 0,1,2,3,4,5,6,7,8,9,10,11,12,13,14,15);
            acc0 = __builtin_amdgcn_wmma_f32_16x16x32_f16(false, a[s], false, b0,
                                                          (short)0, acc0, false, false);
            acc1 = __builtin_amdgcn_wmma_f32_16x16x32_f16(false, a[s], false, b1,
                                                          (short)0, acc1, false, false);
        }

        // -------- spill C tiles to wave-private LDS (stride 36 -> conflict-free) --------
        // C/D layout: lane l: N=l%16 ; VGPR r: M = r + 8*(l>=16)
        #pragma unroll
        for (int r = 0; r < 8; ++r) {
            const int row = r + 8 * hi;
            wout[row * 36 + m]      = acc0[r];
            wout[row * 36 + 16 + m] = acc1[r];
        }

        // -------- extract 9 diagonals: out[m, dx] = band[m][m+dx+4] --------
        // lanes<16 handle even dx {-4,-2,0,2,4}, lanes>=16 odd dx {-3,-1,1,3}
        #pragma unroll
        for (int q = 0; q < 5; ++q) {
            const int dx = 2 * q + hi - PAD_;
            if (dx <= PAD_) {
                const float v = wout[m * 36 + (m + dx + PAD_)] * inv_c;
                const int ch = (dy + PAD_) * 9 + (dx + PAD_);
                out[((size_t)b * 81 + ch) * HW_ + (size_t)y * W_ + (xw + m)] = v;
            }
        }
    }
}

extern "C" void kernel_launch(void* const* d_in, const int* in_sizes, int n_in,
                              void* d_out, int out_size, void* d_ws, size_t ws_size,
                              hipStream_t stream)
{
    (void)in_sizes; (void)n_in; (void)d_ws; (void)ws_size; (void)out_size;
    const float* first  = (const float*)d_in[0];
    const float* second = (const float*)d_in[1];
    float* out = (float*)d_out;

    dim3 grid(W_ / XTILE, H_, B_);   // (4, 128, 4) = 2048 blocks
    dim3 block(128);                 // 4 waves
    corr81_wmma_kernel<<<grid, block, 0, stream>>>(first, second, out);
}
